// Label_Histogram_89163521065176
// MI455X (gfx1250) — matrix operand order repaired and checked by reading, not compiled, not run on hardware
//
#include <hip/hip_runtime.h>
#include <hip/hip_bf16.h>

typedef __attribute__((ext_vector_type(16))) _Float16 v16h;
typedef __attribute__((ext_vector_type(2)))  _Float16 h2;
typedef __attribute__((ext_vector_type(8)))  float    v8f;

#define HEADS_ 4
#define KCLUS_ 8
#define TAU_   10.0f
#define NFRAG  (16*4*32*8)   // 64 fragments x 32 lanes x 8 dwords (packed f16 pairs)

// ---------------------------------------------------------------------------
// Kernel 0: pack W [512,64] f32 -> B-fragment-ordered f16 dword pairs.
// B (32x16 f16, 8 VGPRs), contiguous K order chosen here: lanes 0-15 hold K=0..15 (VGPR v: K=2v,2v+1),
// lanes 16-31 hold K=16..31.
// LDS layout per fragment f = kb*4+nt: dword v of lane L stored at
//   f*256 + (v>>2)*128 + L*4 + (v&3)
// so each half-fragment is one ds_load_b128 with 4-dword lane stride
// (minimum-conflict pattern for 16B/lane across 64 banks).
// ---------------------------------------------------------------------------
__global__ void pack_W_kernel(const float* __restrict__ W, unsigned* __restrict__ wfrag) {
    int t = blockIdx.x * blockDim.x + threadIdx.x;
    if (t >= NFRAG) return;
    int v    = t & 7;
    int lane = (t >> 3) & 31;
    int nt   = (t >> 8) & 3;
    int kb   = t >> 10;
    int n    = nt * 16 + (lane & 15);
    int kl   = (lane >> 4) * 16 + v * 2;
    int row  = kb * 32 + kl;
    union { _Float16 h[2]; unsigned u; } p;
    p.h[0] = (_Float16)W[row * 64 + n];
    p.h[1] = (_Float16)W[(row + 1) * 64 + n];
    int f = kb * 4 + nt;
    wfrag[f * 256 + (v >> 2) * 128 + lane * 4 + (v & 3)] = p.u;
}

// ---------------------------------------------------------------------------
// Kernel 1: per-edge cluster weights w[E,8].
// x = x_j + e_ij; dist_hk = ||k_hk - x||^2; t = 1/(1+dist) (gamma=1);
// per-head normalize *TAU, mean over heads, softmax over K.
// ---------------------------------------------------------------------------
__global__ void edge_weights_kernel(const float* __restrict__ xj,
                                    const float* __restrict__ eij,
                                    const float* __restrict__ cent,
                                    float* __restrict__ wbuf, int E) {
    __shared__ float sc[HEADS_ * KCLUS_ * 64];       // 8 KB of cluster centers
    int tid = threadIdx.x;
    for (int i = tid; i < HEADS_ * KCLUS_ * 64; i += blockDim.x) sc[i] = cent[i];
    __syncthreads();

    int e = blockIdx.x * blockDim.x + tid;
    if (e >= E) return;

    float x[64];
    const float4* a4 = (const float4*)(xj  + (size_t)e * 64);
    const float4* b4 = (const float4*)(eij + (size_t)e * 64);
#pragma unroll
    for (int q = 0; q < 16; ++q) {
        float4 va = a4[q], vb = b4[q];
        x[4*q+0] = va.x + vb.x; x[4*q+1] = va.y + vb.y;
        x[4*q+2] = va.z + vb.z; x[4*q+3] = va.w + vb.w;
    }

    float acc[8];
#pragma unroll
    for (int k = 0; k < 8; ++k) acc[k] = 0.f;

    for (int h = 0; h < HEADS_; ++h) {
        float t[8]; float s = 0.f;
        for (int k = 0; k < KCLUS_; ++k) {
            const float4* kf4 = (const float4*)&sc[(h * 8 + k) * 64];  // broadcast b128
            float d = 0.f;
#pragma unroll
            for (int q = 0; q < 16; ++q) {
                float4 kv = kf4[q];
                float d0 = kv.x - x[4*q+0];
                float d1 = kv.y - x[4*q+1];
                float d2 = kv.z - x[4*q+2];
                float d3 = kv.w - x[4*q+3];
                d = fmaf(d0, d0, d); d = fmaf(d1, d1, d);
                d = fmaf(d2, d2, d); d = fmaf(d3, d3, d);
            }
            float tt = 1.0f / (1.0f + d);
            t[k] = tt; s += tt;
        }
        float scl = (TAU_ * (1.0f / HEADS_)) / s;
#pragma unroll
        for (int k = 0; k < 8; ++k) acc[k] += t[k] * scl;
    }

    float m = acc[0];
#pragma unroll
    for (int k = 1; k < 8; ++k) m = fmaxf(m, acc[k]);
    float ex[8]; float sum = 0.f;
#pragma unroll
    for (int k = 0; k < 8; ++k) { ex[k] = __expf(acc[k] - m); sum += ex[k]; }
    float inv = 1.0f / sum;
    float4 o0 = make_float4(ex[0]*inv, ex[1]*inv, ex[2]*inv, ex[3]*inv);
    float4 o1 = make_float4(ex[4]*inv, ex[5]*inv, ex[6]*inv, ex[7]*inv);
    float4* wp = (float4*)(wbuf + (size_t)e * 8);
    wp[0] = o0; wp[1] = o1;
}

// ---------------------------------------------------------------------------
// Kernel 2: per-16-edge tile: Y[e,:] = (w[e]⊗msg[e]) @ W via WMMA f16 16x16x32,
// K=512 (16 k-steps, each k-step inside one cluster block), then atomic
// scatter Y into out[index[e]*64 + :].
// ---------------------------------------------------------------------------
__global__ void __launch_bounds__(256, 1)
wmma_scatter_kernel(const float* __restrict__ msg,
                    const float* __restrict__ wbuf,
                    const int* __restrict__ index,
                    const unsigned* __restrict__ wfrag,
                    float* __restrict__ out, int E, int nTiles) {
    __shared__ unsigned sB[NFRAG];                    // 64 KB: W in fragment order
    int tid = threadIdx.x;
    for (int i = tid; i < NFRAG; i += 256) sB[i] = wfrag[i];
    __syncthreads();

    const int lane  = tid & 31;
    const int wave  = tid >> 5;
    const int khalf = lane >> 4;      // A-layout K-half (0-15 vs 16-31)
    const int erow  = lane & 15;      // A-matrix row M
    int gwave  = blockIdx.x * 8 + wave;
    int nWaves = gridDim.x * 8;

    for (int tile = gwave; tile < nTiles; tile += nWaves) {
        int e0 = tile * 16;
        int e  = e0 + erow;
        bool ein = (e < E);
        int ec = ein ? e : (E - 1);

        // msg values this lane needs (cols p*32 + khalf*16 + hi*8 + {0..7}),
        // pre-converted to packed f16 pairs: 16 VGPRs total.
        h2 mr[2][2][4];
        const float4* mp = (const float4*)(msg + (size_t)ec * 64);
#pragma unroll
        for (int p = 0; p < 2; ++p)
#pragma unroll
            for (int hi = 0; hi < 2; ++hi) {
                int f4 = p * 8 + khalf * 4 + hi * 2;
                float4 u0 = mp[f4], u1 = mp[f4 + 1];
                mr[p][hi][0] = h2{(_Float16)u0.x, (_Float16)u0.y};
                mr[p][hi][1] = h2{(_Float16)u0.z, (_Float16)u0.w};
                mr[p][hi][2] = h2{(_Float16)u1.x, (_Float16)u1.y};
                mr[p][hi][3] = h2{(_Float16)u1.z, (_Float16)u1.w};
            }

        // cluster weights as f16 broadcast pairs
        const float4* wp = (const float4*)(wbuf + (size_t)ec * 8);
        float4 wa = wp[0], wb = wp[1];
        float w8[8] = {wa.x, wa.y, wa.z, wa.w, wb.x, wb.y, wb.z, wb.w};
        h2 ws[8];
#pragma unroll
        for (int k = 0; k < 8; ++k) {
            _Float16 hv = ein ? (_Float16)w8[k] : (_Float16)0.f;
            ws[k] = h2{hv, hv};
        }

        // node indices for the 8 C rows this lane owns (M = khalf*8 + r)
        int idx8[8];
#pragma unroll
        for (int r = 0; r < 8; ++r) {
            int ee = e0 + khalf * 8 + r;
            idx8[r] = (ee < E) ? index[ee] : -1;
        }

        v8f acc0 = {}, acc1 = {}, acc2 = {}, acc3 = {};

#pragma unroll
        for (int kb = 0; kb < 16; ++kb) {
            // compiler memory barrier: keep the LDS B-fragment loads inside the
            // loop (LICM hoisting them caused scratch spills)
            asm volatile("" ::: "memory");
            const int p = kb & 1;        // msg column half within cluster block
            const int c = kb >> 1;       // cluster for this 32-wide k-step
            h2 s2 = ws[c];
            union { h2 p8[8]; v16h v; } a;
#pragma unroll
            for (int v = 0; v < 8; ++v)
                a.p8[v] = mr[p][v >> 2][v & 3] * s2;   // v_pk_mul_f16

#pragma unroll
            for (int nt = 0; nt < 4; ++nt) {
                int f = kb * 4 + nt;
                union { uint4 q[2]; v16h v; } b;
                b.q[0] = *(const uint4*)&sB[f * 256 + lane * 4];        // ds_load_b128
                b.q[1] = *(const uint4*)&sB[f * 256 + 128 + lane * 4];  // ds_load_b128
                if (nt == 0)
                    acc0 = __builtin_amdgcn_wmma_f32_16x16x32_f16(false, a.v, false, b.v, (short)0, acc0, false, false);
                else if (nt == 1)
                    acc1 = __builtin_amdgcn_wmma_f32_16x16x32_f16(false, a.v, false, b.v, (short)0, acc1, false, false);
                else if (nt == 2)
                    acc2 = __builtin_amdgcn_wmma_f32_16x16x32_f16(false, a.v, false, b.v, (short)0, acc2, false, false);
                else
                    acc3 = __builtin_amdgcn_wmma_f32_16x16x32_f16(false, a.v, false, b.v, (short)0, acc3, false, false);
            }
        }

        // C layout: lane -> N = lane&15 (rows +8 for lanes 16-31), VGPR r -> M
        int col = lane & 15;
#pragma unroll
        for (int r = 0; r < 8; ++r) {
            int node = idx8[r];
            if (node >= 0) {
                float* ob = out + (size_t)node * 64 + col;
                atomicAdd(ob +  0, acc0[r]);
                atomicAdd(ob + 16, acc1[r]);
                atomicAdd(ob + 32, acc2[r]);
                atomicAdd(ob + 48, acc3[r]);
            }
        }
    }
}

// ---------------------------------------------------------------------------
// Kernel 3: out = leaky_relu(out + b, 0.01) in place.
// ---------------------------------------------------------------------------
__global__ void finalize_kernel(float* __restrict__ out, const float* __restrict__ b, int n) {
    int i = blockIdx.x * blockDim.x + threadIdx.x;
    if (i >= n) return;
    float v = out[i] + b[i & 63];
    out[i] = v > 0.f ? v : 0.01f * v;
}

extern "C" void kernel_launch(void* const* d_in, const int* in_sizes, int n_in,
                              void* d_out, int out_size, void* d_ws, size_t ws_size,
                              hipStream_t stream) {
    const float* msg  = (const float*)d_in[0];
    // d_in[1] = x_i (unused by the reference math)
    const float* xj   = (const float*)d_in[2];
    const float* eij  = (const float*)d_in[3];
    const int*   idx  = (const int*)d_in[4];
    // d_in[5] = num_nodes scalar (N recoverable from out_size)
    const float* cent = (const float*)d_in[6];
    const float* W    = (const float*)d_in[7];
    const float* bias = (const float*)d_in[8];

    int E = in_sizes[0] / 64;

    unsigned* wfrag = (unsigned*)d_ws;                       // 64 KB
    float*    wbuf  = (float*)((char*)d_ws + NFRAG * 4);     // E*8 floats

    hipMemsetAsync(d_out, 0, (size_t)out_size * sizeof(float), stream);

    pack_W_kernel<<<(NFRAG + 255) / 256, 256, 0, stream>>>(W, wfrag);
    edge_weights_kernel<<<(E + 255) / 256, 256, 0, stream>>>(xj, eij, cent, wbuf, E);

    int nTiles = (E + 15) / 16;
    int blocks = 1024;
    if (blocks * 8 > nTiles) blocks = (nTiles + 7) / 8;
    wmma_scatter_kernel<<<blocks, 256, 0, stream>>>(msg, wbuf, idx, wfrag,
                                                    (float*)d_out, E, nTiles);

    finalize_kernel<<<(out_size + 255) / 256, 256, 0, stream>>>((float*)d_out, bias, out_size);
}
